// CausalSelfAttention_41223096107383
// MI455X (gfx1250) — matrix operand order repaired and checked by reading, not compiled, not run on hardware
//
#include <hip/hip_runtime.h>
#include <hip/hip_bf16.h>

// ---------------------------------------------------------------------------
// Causal self-attention for MI455X (gfx1250), bf16 WMMA + f32 accumulate.
// Pipeline: cvt(f32->bf16) -> QKV GEMMs (WMMA) -> RoPE -> flash-attn (WMMA)
//           -> output GEMM (WMMA, f32 epilogue).
// Uses v_wmma_f32_16x16x32_bf16, global_load_async_to_lds_b128 (ASYNCcnt),
// global_prefetch_b8, and non-temporal final stores.
// ---------------------------------------------------------------------------

#define D_MODEL 2048
#define N_HEADS 16
#define D_HEAD  128
#define SEQ_T   2048
#define BATCH   4
#define M_ROWS  (BATCH * SEQ_T) // 8192

typedef __bf16 bf16_t;
typedef __attribute__((ext_vector_type(16))) bf16_t v16bf;
typedef __attribute__((ext_vector_type(8)))  float  v8f;

union Frag {
    v16bf v;
    unsigned int u[8];
};

__device__ __forceinline__ unsigned short f32_bf16(float f) {
    unsigned int u = __float_as_uint(f);
    u += 0x7FFFu + ((u >> 16) & 1u);           // round-to-nearest-even
    return (unsigned short)(u >> 16);
}
__device__ __forceinline__ float bf16_f32(unsigned short h) {
    return __uint_as_float(((unsigned int)h) << 16);
}
// A-fragment 16x32 bf16, contiguous k order (K = 16*g + 2*j), matching B fills: u32 index (=K/2) for VGPR j, lane-half g.
__device__ __forceinline__ int kpair_idx(int j, int g) {
    return j + 8 * g;
}

// gfx1250 async global->LDS copy (ASYNCcnt); lds_off = LDS byte offset
// (low 32 bits of a generic shared pointer), gaddr = 64-bit global address.
__device__ __forceinline__ void async_load_b128(unsigned lds_off, unsigned long long gaddr) {
    asm volatile("global_load_async_to_lds_b128 %0, %1, off"
                 :: "v"(lds_off), "v"(gaddr) : "memory");
}
__device__ __forceinline__ void wait_asynccnt0() {
    asm volatile("s_wait_asynccnt 0x0" ::: "memory");
}
__device__ __forceinline__ unsigned lds_offset_of(const void* p) {
    return (unsigned)(unsigned long long)p;   // LDS aperture: addr[31:0] = offset
}

// ---------------------------------------------------------------------------
__global__ __launch_bounds__(256)
void cvt_f32_bf16(const float* __restrict__ in, unsigned short* __restrict__ out, int n) {
    int i = blockIdx.x * 256 + threadIdx.x;
    if (i < n) out[i] = f32_bf16(in[i]);
}

// ---------------------------------------------------------------------------
// GEMM: C[M,N] = A[M,K] x B[K,N], bf16 inputs, f32 accum.
// Block tile 64x64, 4 waves (128 thr), K-step 32; 4 WMMAs/wave/iteration.
// EPI=0: write bf16 to [B,H,T,Dh] (QKV head split); EPI=1: write f32 row-major.
template <int EPI>
__global__ __launch_bounds__(128)
void gemm_bf16(const unsigned short* __restrict__ A,
               const unsigned short* __restrict__ B,
               void* __restrict__ out, int Mdim, int Ndim, int Kdim)
{
    __shared__ __align__(16) unsigned int   As[64 * 16];   // [m][k-pair]
    __shared__ __align__(16) unsigned short Bts[64 * 32];  // transposed: [n][k]
    unsigned int* Btu = (unsigned int*)Bts;

    const int tid = threadIdx.x;
    const int w = tid >> 5, lane = tid & 31;
    const int m = lane & 15, g = lane >> 4;
    const int m0 = blockIdx.y * 64, n0 = blockIdx.x * 64;

    v8f acc[4];
#pragma unroll
    for (int i = 0; i < 4; ++i) acc[i] = (v8f){0.f,0.f,0.f,0.f,0.f,0.f,0.f,0.f};

    const unsigned int* Bu = (const unsigned int*)B;
    const int Np = Ndim >> 1;
    const unsigned ldsA = lds_offset_of((const void*)As);
    const unsigned long long Ag = (unsigned long long)(const void*)A;

    for (int k0 = 0; k0 < Kdim; k0 += 32) {
        __syncthreads();
        // A tile 64x32 -> LDS via async DMA (row-major; LDS chunk i == 16B chunk i)
        for (int i = tid; i < 256; i += 128) {
            int r = i >> 2, c = i & 3;
            async_load_b128(ldsA + (unsigned)i * 16,
                            Ag + (((unsigned long long)(m0 + r) * Kdim + k0 + c * 8) << 1));
        }
        // B tile 32x64 -> LDS transposed [n][k]
        for (int i = tid; i < 1024; i += 128) {
            int kk = i >> 5, np = i & 31;
            unsigned int uv = Bu[(size_t)(k0 + kk) * Np + (n0 >> 1) + np];
            int n = np * 2;
            Bts[n * 32 + kk]       = (unsigned short)(uv & 0xFFFFu);
            Bts[(n + 1) * 32 + kk] = (unsigned short)(uv >> 16);
        }
        // prefetch next iteration's tiles into cache while we compute
        if (k0 + 32 < Kdim) {
            __builtin_prefetch((const char*)A + (((size_t)(m0 + (tid >> 2)) * Kdim + k0 + 32) << 1), 0, 0);
            __builtin_prefetch((const char*)B + (((size_t)(k0 + 32 + (tid >> 2)) * Ndim + n0) << 1), 0, 0);
        }
        wait_asynccnt0();
        __syncthreads();

        Frag a;
#pragma unroll
        for (int j = 0; j < 8; ++j)
            a.u[j] = As[(w * 16 + m) * 16 + kpair_idx(j, g)];
        Frag bf[4];
#pragma unroll
        for (int nt = 0; nt < 4; ++nt)
#pragma unroll
            for (int j = 0; j < 8; ++j)
                bf[nt].u[j] = Btu[(nt * 16 + m) * 16 + 8 * g + j];
#pragma unroll
        for (int nt = 0; nt < 4; ++nt)
            acc[nt] = __builtin_amdgcn_wmma_f32_16x16x32_bf16(
                false, a.v, false, bf[nt].v, (short)0, acc[nt], false, false);
    }

#pragma unroll
    for (int nt = 0; nt < 4; ++nt) {
#pragma unroll
        for (int r = 0; r < 8; ++r) {
            int R = m0 + w * 16 + r + 8 * g;     // C/D layout: M = r + 8*g
            int Cc = n0 + nt * 16 + m;           // N = lane 0..15
            float val = acc[nt][r];
            if (EPI == 0) {
                int b = R >> 11, t = R & (SEQ_T - 1);
                int h = Cc >> 7, d = Cc & (D_HEAD - 1);
                ((unsigned short*)out)[((((size_t)b * N_HEADS + h) * SEQ_T + t) << 7) + d]
                    = f32_bf16(val);
            } else {
                __builtin_nontemporal_store(val, &((float*)out)[(size_t)R * Ndim + Cc]);
            }
        }
    }
}

// ---------------------------------------------------------------------------
// RoPE on q,k bf16 [B,H,T,Dh] in-place. One thread per (row, j<64) pair.
__global__ __launch_bounds__(256)
void rope_kernel(unsigned short* __restrict__ q, unsigned short* __restrict__ k, int total)
{
    int i = blockIdx.x * 256 + threadIdx.x;
    if (i >= total) return;
    int j = i & 63;
    int row = i >> 6;                 // [B,H,T] flat
    int t = row & (SEQ_T - 1);
    float invf = __powf(10000.0f, -(float)j * (1.0f / 64.0f));
    float sa, ca;
    __sincosf((float)t * invf, &sa, &ca);
    size_t base = ((size_t)row << 7) + j;
    {
        float x1 = bf16_f32(q[base]), x2 = bf16_f32(q[base + 64]);
        q[base]      = f32_bf16(x1 * ca - x2 * sa);
        q[base + 64] = f32_bf16(x2 * ca + x1 * sa);
    }
    {
        float x1 = bf16_f32(k[base]), x2 = bf16_f32(k[base + 64]);
        k[base]      = f32_bf16(x1 * ca - x2 * sa);
        k[base + 64] = f32_bf16(x2 * ca + x1 * sa);
    }
}

// ---------------------------------------------------------------------------
// Flash attention: block = (q-tile 64, head, batch), 4 waves; wave owns 16 rows.
// Streams 32-key K/V tiles; S via 8 WMMAs, O += P*V via 8 WMMAs per tile.
__global__ __launch_bounds__(128)
void attn_kernel(const unsigned short* __restrict__ qg,
                 const unsigned short* __restrict__ kg,
                 const unsigned short* __restrict__ vg,
                 unsigned short* __restrict__ og)
{
    __shared__ __align__(16) unsigned short Ks[32 * 128];   // [key][d]
    __shared__ __align__(16) unsigned short Vts[128 * 32];  // transposed [d][key]
    __shared__ __align__(16) unsigned short Ps[4 * 16 * 32];// per-wave P tile
    unsigned int* Ksu = (unsigned int*)Ks;
    unsigned int* Vtu = (unsigned int*)Vts;
    unsigned int* Psu = (unsigned int*)Ps;

    const int tid = threadIdx.x;
    const int w = tid >> 5, lane = tid & 31;
    const int m = lane & 15, g = lane >> 4;
    const int qt = blockIdx.x, h = blockIdx.y, b = blockIdx.z;
    const int qbase = qt * 64 + w * 16;

    const size_t headoff = (((size_t)b * N_HEADS + h) * SEQ_T) << 7;
    const unsigned int* qgu = (const unsigned int*)(qg + headoff);
    const unsigned int* vgu = (const unsigned int*)(vg + headoff);
    const unsigned long long kgaddr = (unsigned long long)(const void*)(kg + headoff);
    const unsigned ldsK = lds_offset_of((const void*)Ks);

    // Resident Q fragments: 16 rows x 128 (4 chunks of K=32)
    Frag qa[4];
#pragma unroll
    for (int c4 = 0; c4 < 4; ++c4)
#pragma unroll
        for (int j = 0; j < 8; ++j)
            qa[c4].u[j] = qgu[(size_t)(qbase + m) * 64 + c4 * 16 + kpair_idx(j, g)];

    v8f o[8];
#pragma unroll
    for (int i = 0; i < 8; ++i) o[i] = (v8f){0.f,0.f,0.f,0.f,0.f,0.f,0.f,0.f};
    float mrun[8], lrun[8];
#pragma unroll
    for (int r = 0; r < 8; ++r) { mrun[r] = -1e30f; lrun[r] = 0.f; }

    const float scale = 0.08838834764831845f;  // 1/sqrt(128)
    const int nkt = 2 * (qt + 1);

    for (int kt = 0; kt < nkt; ++kt) {
        const int kb = kt * 32;
        __syncthreads();
        // K tile (contiguous 8KB) via async DMA, same layout in LDS
        for (int i = tid; i < 512; i += 128)
            async_load_b128(ldsK + (unsigned)i * 16,
                            kgaddr + ((unsigned long long)kb << 8) + (unsigned long long)i * 16);
        // V tile -> LDS transposed [d][key]
        for (int i = tid; i < 2048; i += 128) {
            int r = i >> 6, p = i & 63;
            unsigned int uv = vgu[(size_t)(kb + r) * 64 + p];
            Vts[(2 * p) * 32 + r]     = (unsigned short)(uv & 0xFFFFu);
            Vts[(2 * p + 1) * 32 + r] = (unsigned short)(uv >> 16);
        }
        if (kt + 1 < nkt) {
            __builtin_prefetch((const char*)kgaddr + ((size_t)(kb + 32) << 8) + tid * 64, 0, 0);
            __builtin_prefetch((const char*)vgu    + ((size_t)(kb + 32) << 8) + tid * 64, 0, 0);
        }
        wait_asynccnt0();
        __syncthreads();

        // Preload all 8 K fragments, then chained WMMAs (loads overlap chain)
        Frag kf[8];
#pragma unroll
        for (int c4 = 0; c4 < 4; ++c4)
#pragma unroll
            for (int s = 0; s < 2; ++s)
#pragma unroll
                for (int j = 0; j < 8; ++j)
                    kf[c4 * 2 + s].u[j] =
                        Ksu[(size_t)(s * 16 + m) * 64 + c4 * 16 + 8 * g + j];

        v8f s0 = (v8f){0.f,0.f,0.f,0.f,0.f,0.f,0.f,0.f};
        v8f s1 = s0;
#pragma unroll
        for (int c4 = 0; c4 < 4; ++c4) {
            s0 = __builtin_amdgcn_wmma_f32_16x16x32_bf16(
                false, qa[c4].v, false, kf[c4 * 2 + 0].v, (short)0, s0, false, false);
            s1 = __builtin_amdgcn_wmma_f32_16x16x32_bf16(
                false, qa[c4].v, false, kf[c4 * 2 + 1].v, (short)0, s1, false, false);
        }

        // scale + causal mask + online softmax (rows split across lane halves)
        float corr[8];
#pragma unroll
        for (int r = 0; r < 8; ++r) {
            int qrow = qbase + r + 8 * g;
            float v0 = s0[r] * scale;
            float v1 = s1[r] * scale;
            if (kb + m > qrow)      v0 = -1e30f;
            if (kb + 16 + m > qrow) v1 = -1e30f;
            float t = fmaxf(v0, v1);
#pragma unroll
            for (int off = 1; off < 16; off <<= 1)
                t = fmaxf(t, __shfl_xor(t, off, 32));
            float nm = fmaxf(mrun[r], t);
            float cr = __expf(mrun[r] - nm);
            float p0 = __expf(v0 - nm);
            float p1 = __expf(v1 - nm);
            float rs = p0 + p1;
#pragma unroll
            for (int off = 1; off < 16; off <<= 1)
                rs += __shfl_xor(rs, off, 32);
            lrun[r] = lrun[r] * cr + rs;
            mrun[r] = nm;
            corr[r] = cr;
            // stash P (C-layout -> LDS) for re-load in A-fragment layout
            Ps[w * 512 + (r + 8 * g) * 32 + m]      = f32_bf16(p0);
            Ps[w * 512 + (r + 8 * g) * 32 + 16 + m] = f32_bf16(p1);
        }

#pragma unroll
        for (int ct = 0; ct < 8; ++ct)
#pragma unroll
            for (int r = 0; r < 8; ++r)
                o[ct][r] *= corr[r];

        // O += P (16x32) x V (32x128); per-wave LDS is in-order within wave
        Frag pf;
#pragma unroll
        for (int j = 0; j < 8; ++j)
            pf.u[j] = Psu[w * 256 + m * 16 + kpair_idx(j, g)];
#pragma unroll
        for (int ct = 0; ct < 8; ++ct) {
            Frag vf;
#pragma unroll
            for (int j = 0; j < 8; ++j)
                vf.u[j] = Vtu[(ct * 16 + m) * 16 + 8 * g + j];
            o[ct] = __builtin_amdgcn_wmma_f32_16x16x32_bf16(
                false, pf.v, false, vf.v, (short)0, o[ct], false, false);
        }
    }

    // normalize + write bf16 [B,T,H,Dh] (== concat-head row-major for Wo GEMM)
#pragma unroll
    for (int r = 0; r < 8; ++r) {
        float inv = (lrun[r] > 0.f) ? 1.0f / lrun[r] : 0.0f;
        int qrow = qbase + r + 8 * g;
        size_t outb = (((size_t)b * SEQ_T + qrow) * N_HEADS + h) << 7;
#pragma unroll
        for (int ct = 0; ct < 8; ++ct)
            og[outb + ct * 16 + m] = f32_bf16(o[ct][r] * inv);
    }
}

// ---------------------------------------------------------------------------
extern "C" void kernel_launch(void* const* d_in, const int* in_sizes, int n_in,
                              void* d_out, int out_size, void* d_ws, size_t ws_size,
                              hipStream_t stream) {
    (void)in_sizes; (void)n_in; (void)out_size; (void)ws_size;
    const float* x  = (const float*)d_in[0];
    const float* wq = (const float*)d_in[1];
    const float* wk = (const float*)d_in[2];
    const float* wv = (const float*)d_in[3];
    const float* wo = (const float*)d_in[4];

    const size_t MC = (size_t)M_ROWS * D_MODEL;   // 16,777,216 elems
    const size_t CC = (size_t)D_MODEL * D_MODEL;  //  4,194,304 elems

    unsigned short* p = (unsigned short*)d_ws;    // ~168 MB total bf16 scratch
    unsigned short* xb  = p; p += MC;             // x bf16; later reused as attn out
    unsigned short* wqb = p; p += CC;
    unsigned short* wkb = p; p += CC;
    unsigned short* wvb = p; p += CC;
    unsigned short* wob = p; p += CC;
    unsigned short* qh  = p; p += MC;             // [B,H,T,Dh]
    unsigned short* kh  = p; p += MC;
    unsigned short* vh  = p; p += MC;

    cvt_f32_bf16<<<(int)((MC + 255) / 256), 256, 0, stream>>>(x,  xb,  (int)MC);
    cvt_f32_bf16<<<(int)((CC + 255) / 256), 256, 0, stream>>>(wq, wqb, (int)CC);
    cvt_f32_bf16<<<(int)((CC + 255) / 256), 256, 0, stream>>>(wk, wkb, (int)CC);
    cvt_f32_bf16<<<(int)((CC + 255) / 256), 256, 0, stream>>>(wv, wvb, (int)CC);
    cvt_f32_bf16<<<(int)((CC + 255) / 256), 256, 0, stream>>>(wo, wob, (int)CC);

    dim3 gg(D_MODEL / 64, M_ROWS / 64);
    gemm_bf16<0><<<gg, 128, 0, stream>>>(xb, wqb, qh, M_ROWS, D_MODEL, D_MODEL);
    gemm_bf16<0><<<gg, 128, 0, stream>>>(xb, wkb, kh, M_ROWS, D_MODEL, D_MODEL);
    gemm_bf16<0><<<gg, 128, 0, stream>>>(xb, wvb, vh, M_ROWS, D_MODEL, D_MODEL);

    int ropeN = BATCH * N_HEADS * SEQ_T * 64;     // 8,388,608
    rope_kernel<<<ropeN / 256, 256, 0, stream>>>(qh, kh, ropeN);

    dim3 ag(SEQ_T / 64, N_HEADS, BATCH);
    attn_kernel<<<ag, 128, 0, stream>>>(qh, kh, vh, xb);

    gemm_bf16<1><<<gg, 128, 0, stream>>>(xb, wob, d_out, M_ROWS, D_MODEL, D_MODEL);
}
